// MyModel_87522843561254
// MI455X (gfx1250) — matrix operand order repaired and checked by reading, not compiled, not run on hardware
//
#include <hip/hip_runtime.h>

typedef __attribute__((ext_vector_type(16))) _Float16 v16h;
typedef __attribute__((ext_vector_type(8)))  float    v8f;

#define TT 512
#define FF 32
#define HH 15
#define H3 45
#define XS 20               // padded LDS column stride (floats): 16B aligned, conflict-spread
#define BN_EPS 1e-3f

__device__ __forceinline__ float fast_sigmoid(float v) {
    // 1/(1+exp(-v)) with v_rcp_f32 instead of IEEE divide
    return __builtin_amdgcn_rcpf(1.0f + __expf(-v));
}

__launch_bounds__(32)
__global__ void gru_bn_dense_kernel(const float* __restrict__ x,
                                    const float* __restrict__ Wk,    // [32,45]
                                    const float* __restrict__ Uk,    // [15,45]
                                    const float* __restrict__ gamma,
                                    const float* __restrict__ beta,
                                    const float* __restrict__ mean,
                                    const float* __restrict__ var,
                                    const float* __restrict__ pred,  // [15]
                                    float* __restrict__ out)         // [B]
{
    __shared__ float    s_zr[32 * XS];    // (xw+rec) for cols 0..31, col-major [col][row]
    __shared__ float    s_xh[32 * XS];    // xw  for cols 16..47 (index = col-16)
    __shared__ float    s_rh[32 * XS];    // rec for cols 16..47 (index = col-16)
    __shared__ _Float16 s_h16[16 * 16];   // hidden state f16 (A-frag source), [row][k], k=15 kept 0
    __shared__ float    s_hf[16 * 16];    // final f32 hidden state (epilogue only)

    const int l    = threadIdx.x;     // 0..31
    const int col  = l & 15;
    const int hi   = l >> 4;
    const int row  = col;             // batch row within tile for A fragments
    const int kofs = hi * 8;          // A-frag K group offset per ISA layout
    const int tileBase = blockIdx.x * 16;

    // ---- Build B fragments (held in VGPRs for the whole kernel) ----
    // B layout (16-bit, 32x16): lane = N column (l%16), element e = K = 8*(l/16) + e (e<8), 16 + 8*(l/16) + (e-8) (e>=8)
    v16h wb[3], ub[3];
#pragma unroll
    for (int n = 0; n < 3; ++n) {
#pragma unroll
        for (int e = 0; e < 16; ++e) {
            const int K = (e < 8) ? (hi * 8 + e) : (16 + hi * 8 + (e - 8));
            const int N = n * 16 + col;
            float wv = (N < H3) ? Wk[K * H3 + N] : 0.0f;
            float uv = (N < H3 && K < HH) ? Uk[K * H3 + N] : 0.0f;
            wb[n][e] = (_Float16)wv;
            ub[n][e] = (_Float16)uv;
        }
    }

    // ---- Zero hidden state: f16 copy in LDS, f32 copy in registers ----
    // Gate-item mapping is FIXED per lane: row r_ = l&15, j = hi + 2*it (it=0..7).
    float hreg[8];
#pragma unroll
    for (int it = 0; it < 8; ++it) hreg[it] = 0.0f;
    for (int i = l; i < 256; i += 32) s_h16[i] = (_Float16)0.0f;
    __syncthreads();

    // x row base for this lane's A-fragment role:
    // lane<16: K in {0..7, 16..23}; lane>=16: K in {8..15, 24..31}
    const float* xrow = x + (size_t)(tileBase + row) * TT * FF + kofs;

#pragma unroll 1
    for (int t = 0; t < TT; ++t) {
        // -------- Load x A fragment (A layout: elems 0..7 = K kofs.., 8..15 = K 16+kofs..) --------
        const float4* xp = reinterpret_cast<const float4*>(xrow + (size_t)t * FF);
        float4 x0 = xp[0];
        float4 x1 = xp[1];
        float4 x2 = xp[4];   // +16 floats => K = 16 + kofs
        float4 x3 = xp[5];
        __builtin_prefetch(xrow + (size_t)(t + 1) * FF, 0, 3);  // global_prefetch_b8

        v16h ax;
        ax[0]  = (_Float16)x0.x;  ax[1]  = (_Float16)x0.y;
        ax[2]  = (_Float16)x0.z;  ax[3]  = (_Float16)x0.w;
        ax[4]  = (_Float16)x1.x;  ax[5]  = (_Float16)x1.y;
        ax[6]  = (_Float16)x1.z;  ax[7]  = (_Float16)x1.w;
        ax[8]  = (_Float16)x2.x;  ax[9]  = (_Float16)x2.y;
        ax[10] = (_Float16)x2.z;  ax[11] = (_Float16)x2.w;
        ax[12] = (_Float16)x3.x;  ax[13] = (_Float16)x3.y;
        ax[14] = (_Float16)x3.z;  ax[15] = (_Float16)x3.w;

        // -------- Build h A fragment from f16 hidden state (K>=16 are zero pad) --------
        v16h ah = {};
        {
            const _Float16* hrow = &s_h16[row * 16 + kofs];  // 16B-aligned, 8 contiguous halves
#pragma unroll
            for (int e = 0; e < 8; ++e) ah[e] = hrow[e];
        }

        // -------- WMMA: 3 N-tiles each for input projection and recurrent --------
        v8f accx[3], accr[3];
#pragma unroll
        for (int n = 0; n < 3; ++n) {
            v8f z0 = {};
            accx[n] = __builtin_amdgcn_wmma_f32_16x16x32_f16(
                false, ax, false, wb[n], (short)0, z0, false, false);
            v8f z1 = {};
            accr[n] = __builtin_amdgcn_wmma_f32_16x16x32_f16(
                false, ah, false, ub[n], (short)0, z1, false, false);
        }

        // -------- Spill to LDS (D layout: M = v + 8*hi, N = col) --------
        // Tiles 0,1: store xw+rec (z/r gates consume only the sum).
        // Tiles 1,2: store xw and rec separately (h gate needs them apart).
#pragma unroll
        for (int n = 0; n < 2; ++n) {
            float* pz = &s_zr[(n * 16 + col) * XS + kofs];
#pragma unroll
            for (int v = 0; v < 8; ++v) pz[v] = accx[n][v] + accr[n][v];
        }
#pragma unroll
        for (int n = 1; n < 3; ++n) {
            float* px = &s_xh[((n - 1) * 16 + col) * XS + kofs];
            float* pr = &s_rh[((n - 1) * 16 + col) * XS + kofs];
#pragma unroll
            for (int v = 0; v < 8; ++v) { px[v] = accx[n][v]; pr[v] = accr[n][v]; }
        }
        __syncthreads();

        // -------- Gates: branch-free, fixed row per lane, j = hi + 2*it --------
        // Phantom j=15 (lane hi=1, it=7): inputs from zero-padded weight cols are
        // finite; f16 store is forced to 0 so recurrent A-frag K=15 padding stays clean.
#pragma unroll
        for (int it = 0; it < 8; ++it) {
            const int r_ = col;
            const int j  = hi + 2 * it;      // 0..15
            const float zsum = s_zr[ j       * XS + r_];          // col j        (0..15)
            const float rsum = s_zr[(15 + j) * XS + r_];          // col 15+j     (15..30)
            const float xh   = s_xh[(14 + j) * XS + r_];          // col 30+j - 16
            const float rh   = s_rh[(14 + j) * XS + r_];

            const float zg = fast_sigmoid(zsum);
            const float rg = fast_sigmoid(rsum);
            const float pre = xh + rg * rh;
            const float SC = 1.0507009873554805f;
            const float AL = 1.6732632423543772f;
            // branchless SELU: SC*max(pre,0) + SC*AL*(exp(min(pre,0))-1)
            const float hh = SC * fmaxf(pre, 0.0f)
                           + (SC * AL) * (__expf(fminf(pre, 0.0f)) - 1.0f);

            const float hn = hh + zg * (hreg[it] - hh);   // z*h + (1-z)*hh
            hreg[it] = hn;
            s_h16[r_ * 16 + j] = (_Float16)((j == 15) ? 0.0f : hn);
        }
        __syncthreads();
    }

    // -------- Dump register-resident h to LDS for the epilogue --------
#pragma unroll
    for (int it = 0; it < 8; ++it) {
        const int j = hi + 2 * it;
        s_hf[col * 16 + j] = hreg[it];
    }
    __syncthreads();

    // -------- BatchNorm (inference) + Dense(1) --------
    if (l < 16) {
        float acc = 0.0f;
#pragma unroll
        for (int j = 0; j < HH; ++j) {
            const float hv = s_hf[l * 16 + j];
            const float yv = (hv - mean[j]) * rsqrtf(var[j] + BN_EPS) * gamma[j] + beta[j];
            acc += yv * pred[j];
        }
        out[tileBase + l] = acc;
    }
}

extern "C" void kernel_launch(void* const* d_in, const int* in_sizes, int n_in,
                              void* d_out, int out_size, void* d_ws, size_t ws_size,
                              hipStream_t stream) {
    const float* x     = (const float*)d_in[0];
    const float* Wk    = (const float*)d_in[1];
    const float* Uk    = (const float*)d_in[2];
    const float* gamma = (const float*)d_in[3];
    const float* beta  = (const float*)d_in[4];
    const float* mean  = (const float*)d_in[5];
    const float* var   = (const float*)d_in[6];
    const float* pred  = (const float*)d_in[7];
    float* out = (float*)d_out;

    const int B = in_sizes[0] / (TT * FF);   // 4096
    const int grid = B / 16;                 // one wave32 per 16-row tile -> 256 waves
    gru_bn_dense_kernel<<<grid, 32, 0, stream>>>(x, Wk, Uk, gamma, beta, mean, var, pred, out);
}
